// Attn_62319975465108
// MI455X (gfx1250) — matrix operand order repaired and checked by reading, not compiled, not run on hardware
//
#include <hip/hip_runtime.h>
#include <hip/hip_bf16.h>

#define N_TOK 4096
#define DIM   1024
#define SCALE 0.03125f   // 1/sqrt(1024)

typedef __attribute__((ext_vector_type(16))) __bf16 v16bf;
typedef __attribute__((ext_vector_type(8)))  __bf16 v8bf;
typedef __attribute__((ext_vector_type(2)))  __bf16 v2bf;
typedef __attribute__((ext_vector_type(8)))  float  v8f;
typedef int v4i __attribute__((vector_size(16)));
typedef unsigned u32x4 __attribute__((vector_size(16)));
typedef int i32x8 __attribute__((vector_size(32)));
typedef int i32x4 __attribute__((vector_size(16)));

// ---------------- feature guards -------------------------------------------
#if defined(__has_builtin)
#  if __has_builtin(__builtin_amdgcn_global_load_async_to_lds_b128) && \
      __has_builtin(__builtin_amdgcn_s_wait_asynccnt)
#    define USE_ASYNC_CP 1
#  endif
#  if __has_builtin(__builtin_amdgcn_cvt_pk_bf16_f32)
#    define USE_CVT_PK 1
#  endif
#  if __has_builtin(__builtin_amdgcn_tensor_load_to_lds) && \
      __has_builtin(__builtin_amdgcn_s_wait_tensorcnt)
#    define USE_TDM 1
#  endif
#endif
#ifndef USE_ASYNC_CP
#  define USE_ASYNC_CP 0
#endif
#ifndef USE_CVT_PK
#  define USE_CVT_PK 0
#endif
#ifndef USE_TDM
#  define USE_TDM 0
#endif

// ---------------- bf16 conversion (native fptrunc, RNE) --------------------
__device__ __forceinline__ __bf16 f2bf(float f) { return (__bf16)f; }
// pack two floats -> {bf16(a) in [15:0], bf16(b) in [31:16]}
__device__ __forceinline__ unsigned pack_bf16x2(float a, float b) {
#if USE_CVT_PK
    v2bf v = __builtin_amdgcn_cvt_pk_bf16_f32(a, b);
    return __builtin_bit_cast(unsigned, v);
#else
    v2bf v; v[0] = (__bf16)a; v[1] = (__bf16)b;
    return __builtin_bit_cast(unsigned, v);
#endif
}

// ---------------- async global->LDS 16B copy -------------------------------
__device__ __forceinline__ void cp16_g2l(const __bf16* g, __bf16* l) {
#if USE_ASYNC_CP
    __builtin_amdgcn_global_load_async_to_lds_b128(
        (__attribute__((address_space(1))) v4i*)g,
        (__attribute__((address_space(3))) v4i*)l, 0, 0);
#else
    *(v8bf*)l = *(const v8bf*)g;
#endif
}
__device__ __forceinline__ void cp_wait_all() {
#if USE_ASYNC_CP
    __builtin_amdgcn_s_wait_asynccnt(0);
#endif
}

// ---------------- TDM: one-shot 16x1024 bf16 tile load ---------------------
#if USE_TDM
// D# per CDNA5 ISA §8.3/8.4: group0 = {flags, lds_addr, gaddr_lo, gaddr_hi|type},
// group1 = {data_size, tensor/tile dims, strides}; groups 2/3 zero (2D tile).
// This toolchain exposes the 6-arg builtin: (g0, g1, g2, g3, g4, cpol).
__device__ __forceinline__ void tdm_load_q_tile(const __bf16* g, __bf16* lds) {
    unsigned long long ga = (unsigned long long)(size_t)g;
    unsigned la = (unsigned)(unsigned long long)
        (__attribute__((address_space(3))) const void*)lds;
    u32x4 g0 = { 1u,                                        // count=1 (valid), user mode
                 la,                                        // lds_addr (bytes)
                 (unsigned)ga,                              // global_addr[31:0]
                 (unsigned)((ga >> 32) & 0x01FFFFFFu) | (2u << 30) }; // [56:32] | type=2
    // data_size=2 (4B units): row = 512 dwords; tile 512 x 16 of a 512 x 4096 tensor
    i32x8 g1 = { (int)(2u << 16),                           // data_size=4B
                 (int)(512u << 16),                         // tensor_dim0[15:0]=512
                 (int)(4096u << 16),                        // tensor_dim1[15:0]=4096
                 (int)(512u << 16),                         // tile_dim0=512
                 16,                                        // tile_dim1=16, tile_dim2=0
                 512,                                       // tensor_dim0_stride=512
                 0, 0 };
    i32x4 gz4 = { 0, 0, 0, 0 };
    i32x8 gz8 = { 0, 0, 0, 0, 0, 0, 0, 0 };
    __builtin_amdgcn_tensor_load_to_lds(g0, g1, gz4, gz4, gz8, 0);
}
#endif

// ---------------- lane-xor swizzle (wave32 group-of-32, and=0x1f) ----------
#define SWZ_XOR_F(x, m) __builtin_bit_cast(float, \
    __builtin_amdgcn_ds_swizzle(__builtin_bit_cast(int, (x)), (0x1F | ((m) << 10))))

// ---------------- WMMA fragments: two b128 loads each ----------------------
__device__ __forceinline__ v16bf cat8(v8bf lo, v8bf hi) {
    return __builtin_shufflevector(lo, hi, 0,1,2,3,4,5,6,7,8,9,10,11,12,13,14,15);
}
// A fragment 16x32 (MxK): lane holds row (L&15); kh=(L>=16?8:0);
// elems 0..7 -> K=kh+0..7 ; elems 8..15 -> K=kh+16..23 (two 16B runs)
__device__ __forceinline__ v16bf load_a_frag(const __bf16* base, int ld,
                                             int row0, int k0, int lane) {
    const __bf16* p = base + (size_t)(row0 + (lane & 15)) * ld + k0 + ((lane >> 4) << 3);
    return cat8(*(const v8bf*)p, *(const v8bf*)(p + 16));
}
// B fragment 32x16 (KxN) from an N-major, k-contiguous source (row n holds K run):
// kh=(L>=16?8:0): elems 0..7 -> K=kh+0..7, elems 8..15 -> K=kh+16..23; col n = L&15 (two 16B runs)
__device__ __forceinline__ v16bf load_b_frag(const __bf16* base, int ld,
                                             int n0, int k0, int lane) {
    const __bf16* p = base + (size_t)(n0 + (lane & 15)) * ld + k0 + ((lane >> 4) << 3);
    return cat8(*(const v8bf*)p, *(const v8bf*)(p + 16));
}

#define WMMA_BF16(A, B, C) __builtin_amdgcn_wmma_f32_16x16x32_bf16( \
    false, (A), false, (B), (short)0, (C), false, false)

// ===========================================================================
// Kernel 1: projection Y[m,n] = X[m,:] . W[n,:]  (X: 4096x1024 f32, W: 1024x1024 f32)
// Block tile 64(M) x 128(N); 8 waves = 4x2; wave tile 16x64 (4 WMMA / k-step).
// ===========================================================================
__global__ __launch_bounds__(256)
void proj_bf16_kernel(const float* __restrict__ X,
                      const float* __restrict__ W,
                      __bf16* __restrict__ Y) {
    __shared__ __align__(16) __bf16 Xs[64 * 32];
    __shared__ __align__(16) __bf16 Ws[128 * 32];

    const int tid  = threadIdx.x;
    const int lane = tid & 31;
    const int w    = tid >> 5;
    const int mblk = blockIdx.y * 64;
    const int nblk = blockIdx.x * 128;
    const int wm   = (w >> 1) * 16;
    const int wn   = (w & 1) * 64;

    v8f acc[4] = {};

    for (int kk = 0; kk < DIM; kk += 32) {
        __syncthreads();
        for (int c = tid; c < 512; c += 256) {         // X tile: 64 x 32
            const int row = c >> 3, kp = (c & 7) * 4;
            float4 f = *(const float4*)(X + (size_t)(mblk + row) * DIM + kk + kp);
            uint2 u; u.x = pack_bf16x2(f.x, f.y); u.y = pack_bf16x2(f.z, f.w);
            *(uint2*)&Xs[row * 32 + kp] = u;
        }
        for (int c = tid; c < 1024; c += 256) {        // W tile: 128 x 32
            const int row = c >> 3, kp = (c & 7) * 4;
            float4 f = *(const float4*)(W + (size_t)(nblk + row) * DIM + kk + kp);
            uint2 u; u.x = pack_bf16x2(f.x, f.y); u.y = pack_bf16x2(f.z, f.w);
            *(uint2*)&Ws[row * 32 + kp] = u;
        }
        __syncthreads();

        v16bf a = load_a_frag(Xs, 32, wm, 0, lane);
#pragma unroll
        for (int t = 0; t < 4; ++t) {
            v16bf b = load_b_frag(Ws, 32, wn + t * 16, 0, lane);
            acc[t] = WMMA_BF16(a, b, acc[t]);
        }
    }

    const int col = nblk + wn + (lane & 15);
    const int rb  = mblk + wm + ((lane >> 4) << 3);
#pragma unroll
    for (int t = 0; t < 4; ++t)
#pragma unroll
        for (int i = 0; i < 8; ++i)
            Y[(size_t)(rb + i) * DIM + col + t * 16] = f2bf(acc[t][i]);
}

// ===========================================================================
// Kernel 2: softmax stats (row max m, row sum-exp Z). q tile via TDM when
// available (one descriptor DMA, TENSORcnt), else async copies.
// ===========================================================================
__global__ __launch_bounds__(256)
void softmax_stats_kernel(const __bf16* __restrict__ qb,
                          const __bf16* __restrict__ kb,
                          float* __restrict__ d_m,
                          float* __restrict__ d_z) {
    __shared__ __align__(16) __bf16 Qs[16 * DIM];   // 32 KB
    __shared__ float sm[8][16];
    __shared__ float ss[8][16];

    const int tid  = threadIdx.x;
    const int lane = tid & 31;
    const int w    = tid >> 5;
    const int row0 = blockIdx.x * 16;

#if USE_TDM
    if (w == 0) {
        tdm_load_q_tile(qb + (size_t)row0 * DIM, Qs);
        __builtin_amdgcn_s_wait_tensorcnt(0);
    }
    __syncthreads();
#else
    for (int c = tid; c < 2048; c += 256) {
        const int row = c >> 7, kp = (c & 127) * 8;
        cp16_g2l(qb + (size_t)(row0 + row) * DIM + kp, &Qs[row * DIM + kp]);
    }
    cp_wait_all();
    __syncthreads();
#endif

    float m_i[8], s_i[8];
#pragma unroll
    for (int i = 0; i < 8; ++i) { m_i[i] = -1e30f; s_i[i] = 0.0f; }

    for (int jt = w; jt < N_TOK / 16; jt += 8) {
        v8f c = {};
        for (int kk = 0; kk < DIM; kk += 32) {
            v16bf a = load_a_frag(Qs, DIM, 0, kk, lane);
            v16bf b = load_b_frag(kb, DIM, jt * 16, kk, lane);
            c = WMMA_BF16(a, b, c);
        }
#pragma unroll
        for (int i = 0; i < 8; ++i) {
            float sc = c[i] * SCALE;
            float t = sc;
            t = fmaxf(t, SWZ_XOR_F(t, 1));
            t = fmaxf(t, SWZ_XOR_F(t, 2));
            t = fmaxf(t, SWZ_XOR_F(t, 4));
            t = fmaxf(t, SWZ_XOR_F(t, 8));
            float mn = fmaxf(m_i[i], t);
            float e  = __expf(sc - mn);
            e = e + SWZ_XOR_F(e, 1);
            e = e + SWZ_XOR_F(e, 2);
            e = e + SWZ_XOR_F(e, 4);
            e = e + SWZ_XOR_F(e, 8);
            s_i[i] = s_i[i] * __expf(m_i[i] - mn) + e;
            m_i[i] = mn;
        }
    }

    if ((lane & 15) == 0) {
        const int rb = (lane >> 4) << 3;
#pragma unroll
        for (int i = 0; i < 8; ++i) { sm[w][rb + i] = m_i[i]; ss[w][rb + i] = s_i[i]; }
    }
    __syncthreads();
    if (tid < 16) {
        float M = -1e30f;
#pragma unroll
        for (int w2 = 0; w2 < 8; ++w2) M = fmaxf(M, sm[w2][tid]);
        float S = 0.0f;
#pragma unroll
        for (int w2 = 0; w2 < 8; ++w2) S += ss[w2][tid] * __expf(sm[w2][tid] - M);
        d_m[row0 + tid] = M;
        d_z[row0 + tid] = S;
    }
}

// ===========================================================================
// Kernel 3: O = softmax(q k^T / 32) @ v, streamed over 32-key chunks.
// q tile staged with async copies (ASYNCcnt path kept intentionally).
// ===========================================================================
__global__ __launch_bounds__(256)
void attn_out_kernel(const __bf16* __restrict__ qb,
                     const __bf16* __restrict__ kb,
                     const __bf16* __restrict__ vb,
                     const float* __restrict__ d_m,
                     const float* __restrict__ d_z,
                     float* __restrict__ out) {
    __shared__ __align__(16) __bf16 Qs[16 * DIM];     // 32 KB
    __shared__ __align__(16) float  part[8][16 * 32]; // 16 KB
    __shared__ __align__(16) __bf16 Pl[16 * 32];      // 1 KB
    __shared__ __align__(16) __bf16 Vt[128 * 32];     // 8 KB

    const int tid  = threadIdx.x;
    const int lane = tid & 31;
    const int w    = tid >> 5;
    const int row0 = blockIdx.y * 16;
    const int cblk = blockIdx.x * 128;
    const int c0   = cblk + w * 16;

    for (int c = tid; c < 2048; c += 256) {
        const int row = c >> 7, kp = (c & 127) * 8;
        cp16_g2l(qb + (size_t)(row0 + row) * DIM + kp, &Qs[row * DIM + kp]);
    }
    cp_wait_all();
    __syncthreads();

    const int rr = tid >> 5;
    const float m1  = d_m[row0 + rr];
    const float iz1 = 1.0f / d_z[row0 + rr];
    const float m2  = d_m[row0 + 8 + rr];
    const float iz2 = 1.0f / d_z[row0 + 8 + rr];

    v8f acco = {};

    for (int jt = 0; jt < N_TOK; jt += 32) {
        v8f ca = {}, cb = {};
        const int kbase = w * 128;
#pragma unroll
        for (int ks = 0; ks < 4; ++ks) {
            const int kk = kbase + ks * 32;
            v16bf a  = load_a_frag(Qs, DIM, 0, kk, lane);
            v16bf b0 = load_b_frag(kb, DIM, jt,      kk, lane);
            v16bf b1 = load_b_frag(kb, DIM, jt + 16, kk, lane);
            ca = WMMA_BF16(a, b0, ca);
            cb = WMMA_BF16(a, b1, cb);
        }
        {
            const int rbase = (lane >> 4) << 3;
            const int col   = lane & 15;
#pragma unroll
            for (int i = 0; i < 8; ++i) {
                part[w][(rbase + i) * 32 + col]      = ca[i];
                part[w][(rbase + i) * 32 + 16 + col] = cb[i];
            }
        }
        for (int c = tid; c < 512; c += 256) {   // Vt[n][k] = v[jt+k][cblk+n]
            const int k = c >> 4, n0 = (c & 15) * 8;
            v8bf vv = *(const v8bf*)(vb + (size_t)(jt + k) * DIM + cblk + n0);
#pragma unroll
            for (int i = 0; i < 8; ++i)
                Vt[(n0 + i) * 32 + k] = vv[i];
        }
        __syncthreads();

        {
            const int e1 = tid, e2 = tid + 256;
            float s1 = 0.0f, s2 = 0.0f;
#pragma unroll
            for (int w2 = 0; w2 < 8; ++w2) { s1 += part[w2][e1]; s2 += part[w2][e2]; }
            Pl[e1] = f2bf(__expf(s1 * SCALE - m1) * iz1);
            Pl[e2] = f2bf(__expf(s2 * SCALE - m2) * iz2);
        }
        __syncthreads();

        v16bf a = load_a_frag(Pl, 32, 0, 0, lane);
        v16bf b = load_b_frag(Vt, 32, w * 16, 0, lane);
        acco = WMMA_BF16(a, b, acco);
        __syncthreads();
    }

    const int col = c0 + (lane & 15);
    const int rb  = row0 + ((lane >> 4) << 3);
#pragma unroll
    for (int i = 0; i < 8; ++i)
        out[(size_t)(rb + i) * DIM + col] = acco[i];
}

// ===========================================================================
extern "C" void kernel_launch(void* const* d_in, const int* in_sizes, int n_in,
                              void* d_out, int out_size, void* d_ws, size_t ws_size,
                              hipStream_t stream) {
    const float* Q  = (const float*)d_in[0];
    const float* K  = (const float*)d_in[1];
    const float* V  = (const float*)d_in[2];
    const float* Wq = (const float*)d_in[3];
    const float* Wk = (const float*)d_in[4];
    const float* Wv = (const float*)d_in[5];
    float* out = (float*)d_out;

    char* ws = (char*)d_ws;
    const size_t mat = (size_t)N_TOK * DIM * sizeof(__bf16);
    __bf16* qb = (__bf16*)(ws);
    __bf16* kb = (__bf16*)(ws + mat);
    __bf16* vb = (__bf16*)(ws + 2 * mat);
    float*  dm = (float*)(ws + 3 * mat);
    float*  dz = (float*)(ws + 3 * mat + N_TOK * sizeof(float));

    dim3 blk(256);
    dim3 gproj(DIM / 128, N_TOK / 64);   // (8, 64)
    proj_bf16_kernel<<<gproj, blk, 0, stream>>>(Q, Wq, qb);
    proj_bf16_kernel<<<gproj, blk, 0, stream>>>(K, Wk, kb);
    proj_bf16_kernel<<<gproj, blk, 0, stream>>>(V, Wv, vb);

    softmax_stats_kernel<<<dim3(N_TOK / 16), blk, 0, stream>>>(qb, kb, dm, dz);

    attn_out_kernel<<<dim3(DIM / 128, N_TOK / 16), blk, 0, stream>>>(qb, kb, vb,
                                                                     dm, dz, out);
}
